// BigramLanguageModel_76742475645582
// MI455X (gfx1250) — matrix operand order repaired and checked by reading, not compiled, not run on hardware
//
#include <hip/hip_runtime.h>
#include <hip/hip_bf16.h>

// Problem constants
#define Bk 4
#define Tk 2048
#define Ck 512
#define Vk 32000

typedef __attribute__((ext_vector_type(16))) __bf16 v16bf;
typedef __attribute__((ext_vector_type(8)))  __bf16 v8bf;
typedef __attribute__((ext_vector_type(8)))  float  v8f;

// ---------------------------------------------------------------------------
// WMMA helpers (CDNA5 gfx1250, wave32): D = A(16x32 bf16) * B(32x16 bf16) + C
// A layout: lane l (half = l>>4, m = l&15): elem i<8  -> K = 8*half + i
//                                           elem i>=8 -> K = 16 + 8*half + (i-8)
// B layout: lane l (half = l>>4, n = l&15): same K order as A
// C/D layout: lane l: n = l&15, vec r -> m = r + 8*(l>>4)
// ---------------------------------------------------------------------------
__device__ __forceinline__ v8f wmma_bf16(v16bf a, v16bf b, v8f c) {
  return __builtin_amdgcn_wmma_f32_16x16x32_bf16(false, a, false, b, (short)0, c,
                                                 false, false);
}

// A fragment from row-major bf16 matrix (ld = row stride in elements)
__device__ __forceinline__ v16bf load_a_frag_g(const __bf16* __restrict__ base,
                                               int ld, int m0, int k0, int lane) {
  const int half = lane >> 4;
  const int m = m0 + (lane & 15);
  const __bf16* p = base + (size_t)m * ld + k0 + 8 * half;
  v8bf lo = *(const v8bf*)(p);
  v8bf hi = *(const v8bf*)(p + 16);
  v16bf a;
#pragma unroll
  for (int i = 0; i < 8; ++i) { a[i] = lo[i]; a[i + 8] = hi[i]; }
  return a;
}

// B fragment from K-contiguous ("transposed" [N][K]) bf16 matrix, ld = K stride
__device__ __forceinline__ v16bf load_b_frag_g(const __bf16* __restrict__ baseT,
                                               int ld, int k0, int n0, int lane) {
  const int n = n0 + (lane & 15);
  const __bf16* p = baseT + (size_t)n * ld + k0 + 8 * (lane >> 4);
  v8bf lo = *(const v8bf*)(p);
  v8bf hi = *(const v8bf*)(p + 16);
  v16bf b;
#pragma unroll
  for (int i = 0; i < 8; ++i) { b[i] = lo[i]; b[i + 8] = hi[i]; }
  return b;
}

// ---------------------------------------------------------------------------
// Weight convert+transpose: W[K][N] f32 -> Wt[N][K] bf16
// ---------------------------------------------------------------------------
__global__ __launch_bounds__(256) void convert_wT_kernel(
    const float* __restrict__ W, __bf16* __restrict__ Wt, int Kd, int N) {
  size_t i = (size_t)blockIdx.x * 256 + threadIdx.x;
  if (i >= (size_t)Kd * N) return;
  unsigned n = (unsigned)(i / (unsigned)Kd);
  unsigned kk = (unsigned)(i % (unsigned)Kd);
  Wt[i] = (__bf16)W[(size_t)kk * N + n];
}

// ---------------------------------------------------------------------------
// Embedding: x[b,t,c] = bf16(token_emb[idx[b,t]][c] + pos_emb[t][c])
// ---------------------------------------------------------------------------
__global__ __launch_bounds__(256) void embed_kernel(
    const int* __restrict__ idx, const float* __restrict__ tok,
    const float* __restrict__ pos, __bf16* __restrict__ x) {
  size_t i = (size_t)blockIdx.x * 256 + threadIdx.x;
  if (i >= (size_t)Bk * Tk * Ck) return;
  int c = (int)(i & (Ck - 1));
  int bt = (int)(i >> 9);          // /C
  int t = bt & (Tk - 1);
  int token = idx[bt];
  x[i] = (__bf16)(tok[(size_t)token * Ck + c] + pos[(size_t)t * Ck + c]);
}

// ---------------------------------------------------------------------------
// Generic GEMM: out[M][N] bf16 row-major = A[M][K] bf16 @ Wt[N][K] bf16
// 8 waves / block; each wave: one 16x64 tile (4 WMMA n-tiles, shared A frag)
// ---------------------------------------------------------------------------
__global__ __launch_bounds__(256) void gemm_rm_kernel(
    const __bf16* __restrict__ A, const __bf16* __restrict__ Wt,
    __bf16* __restrict__ out, int M, int Kd, int N) {
  int wv = (int)((blockIdx.x * 256 + threadIdx.x) >> 5);
  int lane = threadIdx.x & 31;
  int ntiles = N >> 6;
  int mt = wv / ntiles, nb = wv % ntiles;
  if (mt >= (M >> 4)) return;
  int m0 = mt << 4, n0 = nb << 6;
  v8f a0 = {}, a1 = {}, a2 = {}, a3 = {};
  for (int k0 = 0; k0 < Kd; k0 += 32) {
    v16bf af = load_a_frag_g(A, Kd, m0, k0, lane);
    v16bf b0 = load_b_frag_g(Wt, Kd, k0, n0, lane);
    v16bf b1 = load_b_frag_g(Wt, Kd, k0, n0 + 16, lane);
    v16bf b2 = load_b_frag_g(Wt, Kd, k0, n0 + 32, lane);
    v16bf b3 = load_b_frag_g(Wt, Kd, k0, n0 + 48, lane);
    a0 = wmma_bf16(af, b0, a0);
    a1 = wmma_bf16(af, b1, a1);
    a2 = wmma_bf16(af, b2, a2);
    a3 = wmma_bf16(af, b3, a3);
  }
  int n = lane & 15, mb = m0 + 8 * (lane >> 4);
#pragma unroll
  for (int r = 0; r < 8; ++r) {
    size_t row = (size_t)(mb + r) * N;
    out[row + n0 + n]      = (__bf16)a0[r];
    out[row + n0 + 16 + n] = (__bf16)a1[r];
    out[row + n0 + 32 + n] = (__bf16)a2[r];
    out[row + n0 + 48 + n] = (__bf16)a3[r];
  }
}

// ---------------------------------------------------------------------------
// V projection with transposed store: vt[b][c][t] = (x @ WvT)[b*T+t][c]
// Makes P@V B-fragments K-contiguous.
// ---------------------------------------------------------------------------
__global__ __launch_bounds__(256) void gemm_vt_kernel(
    const __bf16* __restrict__ A, const __bf16* __restrict__ Wt,
    __bf16* __restrict__ vt) {
  const int M = Bk * Tk, Kd = Ck, N = Ck;
  int wv = (int)((blockIdx.x * 256 + threadIdx.x) >> 5);
  int lane = threadIdx.x & 31;
  int ntiles = N >> 6;
  int mt = wv / ntiles, nb = wv % ntiles;
  if (mt >= (M >> 4)) return;
  int m0 = mt << 4, n0 = nb << 6;
  v8f a0 = {}, a1 = {}, a2 = {}, a3 = {};
  for (int k0 = 0; k0 < Kd; k0 += 32) {
    v16bf af = load_a_frag_g(A, Kd, m0, k0, lane);
    v16bf b0 = load_b_frag_g(Wt, Kd, k0, n0, lane);
    v16bf b1 = load_b_frag_g(Wt, Kd, k0, n0 + 16, lane);
    v16bf b2 = load_b_frag_g(Wt, Kd, k0, n0 + 32, lane);
    v16bf b3 = load_b_frag_g(Wt, Kd, k0, n0 + 48, lane);
    a0 = wmma_bf16(af, b0, a0);
    a1 = wmma_bf16(af, b1, a1);
    a2 = wmma_bf16(af, b2, a2);
    a3 = wmma_bf16(af, b3, a3);
  }
  int n = lane & 15, mb = m0 + 8 * (lane >> 4);
#pragma unroll
  for (int r = 0; r < 8; ++r) {
    int mg = mb + r;
    int bb = mg >> 11;               // /T
    int tt = mg & (Tk - 1);
    size_t base = ((size_t)bb * Ck) * Tk + tt;
    vt[base + (size_t)(n0 + n) * Tk]      = (__bf16)a0[r];
    vt[base + (size_t)(n0 + 16 + n) * Tk] = (__bf16)a1[r];
    vt[base + (size_t)(n0 + 32 + n) * Tk] = (__bf16)a2[r];
    vt[base + (size_t)(n0 + 48 + n) * Tk] = (__bf16)a3[r];
  }
}

// ---------------------------------------------------------------------------
// Logits GEMM: out f32 = attn[8192][512] @ WoT[32000][512] + bo, store to d_out
// ---------------------------------------------------------------------------
__global__ __launch_bounds__(256) void gemm_logits_kernel(
    const __bf16* __restrict__ A, const __bf16* __restrict__ Wt,
    const float* __restrict__ bo, float* __restrict__ out) {
  const int M = Bk * Tk, Kd = Ck, N = Vk;
  int wv = (int)((blockIdx.x * 256 + threadIdx.x) >> 5);
  int lane = threadIdx.x & 31;
  int ntiles = N >> 6;                 // 500
  int mt = wv / ntiles, nb = wv % ntiles;
  if (mt >= (M >> 4)) return;
  int m0 = mt << 4, n0 = nb << 6;
  v8f a0 = {}, a1 = {}, a2 = {}, a3 = {};
  for (int k0 = 0; k0 < Kd; k0 += 32) {
    v16bf af = load_a_frag_g(A, Kd, m0, k0, lane);
    v16bf b0 = load_b_frag_g(Wt, Kd, k0, n0, lane);
    v16bf b1 = load_b_frag_g(Wt, Kd, k0, n0 + 16, lane);
    v16bf b2 = load_b_frag_g(Wt, Kd, k0, n0 + 32, lane);
    v16bf b3 = load_b_frag_g(Wt, Kd, k0, n0 + 48, lane);
    a0 = wmma_bf16(af, b0, a0);
    a1 = wmma_bf16(af, b1, a1);
    a2 = wmma_bf16(af, b2, a2);
    a3 = wmma_bf16(af, b3, a3);
  }
  int n = lane & 15, mb = m0 + 8 * (lane >> 4);
  float c0 = bo[n0 + n], c1 = bo[n0 + 16 + n], c2 = bo[n0 + 32 + n], c3 = bo[n0 + 48 + n];
#pragma unroll
  for (int r = 0; r < 8; ++r) {
    size_t row = (size_t)(mb + r) * N;
    out[row + n0 + n]      = a0[r] + c0;
    out[row + n0 + 16 + n] = a1[r] + c1;
    out[row + n0 + 32 + n] = a2[r] + c2;
    out[row + n0 + 48 + n] = a3[r] + c3;
  }
}

// ---------------------------------------------------------------------------
// Causal attention, one 16-query tile per workgroup (8 waves).
// Full 16 x T f32 score block in LDS (131 KB of the WGP's 320 KB).
// ---------------------------------------------------------------------------
__global__ __launch_bounds__(256) void attn_kernel(
    const __bf16* __restrict__ q, const __bf16* __restrict__ k,
    const __bf16* __restrict__ vt, __bf16* __restrict__ attn) {
  constexpr int SROW = Tk + 1;                 // f32 row stride: bank-conflict free
  __shared__ float S[16 * SROW];               // 131136 bytes
  __shared__ float redM[16 * 17];
  __shared__ float redS[16 * 17];
  __shared__ float rowmax[16];
  __shared__ float rowrinv[16];

  const int b = blockIdx.x >> 7;               // T/16 = 128 query tiles per batch
  const int qt = blockIdx.x & 127;
  const int m0 = qt << 4;
  const int wave = threadIdx.x >> 5;
  const int lane = threadIdx.x & 31;
  const int tid = threadIdx.x;
  const int stiles = qt + 1;                   // visible key tiles (causal)
  const int nvis = stiles << 4;

  const __bf16* qb = q + (size_t)b * Tk * Ck;
  const __bf16* kb = k + (size_t)b * Tk * Ck;
  const __bf16* vb = vt + (size_t)b * Ck * Tk;

  // Preload this wave's 16 q A-fragments (K = C = 512 -> 16 chunks of 32)
  v16bf qa[16];
#pragma unroll
  for (int kc = 0; kc < 16; ++kc) qa[kc] = load_a_frag_g(qb, Ck, m0, kc * 32, lane);

  const float scale = 0.04419417382415922f;    // 512^-0.5

  // Phase 1: S = scale * q @ k^T with causal mask; key tiles striped over waves.
  // k row-major [T][C] is exactly the K-contiguous B layout for q@k^T.
  for (int s = wave; s < stiles; s += 8) {
    v8f acc = {};
#pragma unroll
    for (int kc = 0; kc < 16; ++kc) {
      v16bf bf = load_b_frag_g(kb, Ck, kc * 32, s * 16, lane);
      acc = wmma_bf16(qa[kc], bf, acc);
    }
    int n = lane & 15, mb = 8 * (lane >> 4);
    int key = s * 16 + n;
#pragma unroll
    for (int r = 0; r < 8; ++r) {
      int m = mb + r;
      float v = acc[r] * scale;
      if (key > m0 + m) v = -3.0e38f;          // causal mask
      S[m * SROW + key] = v;
    }
  }
  __syncthreads();

  // Phase 2: row max then row sum-of-exp (16 threads per row, f32)
  {
    int r = tid >> 4, c = tid & 15;
    float mx = -3.0e38f;
    for (int key = c; key < nvis; key += 16) mx = fmaxf(mx, S[r * SROW + key]);
    redM[r * 17 + c] = mx;
  }
  __syncthreads();
  if (tid < 16) {
    float mx = redM[tid * 17];
#pragma unroll
    for (int c = 1; c < 16; ++c) mx = fmaxf(mx, redM[tid * 17 + c]);
    rowmax[tid] = mx;
  }
  __syncthreads();
  {
    int r = tid >> 4, c = tid & 15;
    float mx = rowmax[r], sm = 0.f;
    for (int key = c; key < nvis; key += 16) sm += __expf(S[r * SROW + key] - mx);
    redS[r * 17 + c] = sm;
  }
  __syncthreads();
  if (tid < 16) {
    float sm = 0.f;
#pragma unroll
    for (int c = 0; c < 16; ++c) sm += redS[tid * 17 + c];
    rowrinv[tid] = 1.0f / sm;
  }
  __syncthreads();

  // Phase 3: attn = P @ V. Each wave owns a 64-channel chunk; A-fragments are
  // built from LDS scores with exp-normalize on the fly (bf16), B from vt.
  const int c0 = wave * 64;
  const int nk32 = (nvis + 31) & ~31;
  v8f o0 = {}, o1 = {}, o2 = {}, o3 = {};
  const int half = lane >> 4;
  const int mrow = lane & 15;
  const float mx = rowmax[mrow];
  const float ri = rowrinv[mrow];
  for (int kk = 0; kk < nk32; kk += 32) {
    v16bf pa;
#pragma unroll
    for (int i = 0; i < 8; ++i) {
      int key0 = kk + 8 * half + i;
      int key1 = kk + 16 + 8 * half + i;
      float p0 = (key0 < nvis) ? __expf(S[mrow * SROW + key0] - mx) * ri : 0.f;
      float p1 = (key1 < nvis) ? __expf(S[mrow * SROW + key1] - mx) * ri : 0.f;
      pa[i] = (__bf16)p0;
      pa[i + 8] = (__bf16)p1;
    }
    v16bf b0 = load_b_frag_g(vb, Tk, kk, c0, lane);
    v16bf b1 = load_b_frag_g(vb, Tk, kk, c0 + 16, lane);
    v16bf b2 = load_b_frag_g(vb, Tk, kk, c0 + 32, lane);
    v16bf b3 = load_b_frag_g(vb, Tk, kk, c0 + 48, lane);
    o0 = wmma_bf16(pa, b0, o0);
    o1 = wmma_bf16(pa, b1, o1);
    o2 = wmma_bf16(pa, b2, o2);
    o3 = wmma_bf16(pa, b3, o3);
  }
  int n = lane & 15, mb2 = 8 * (lane >> 4);
#pragma unroll
  for (int r = 0; r < 8; ++r) {
    size_t row = ((size_t)b * Tk + m0 + mb2 + r) * Ck;
    attn[row + c0 + n]      = (__bf16)o0[r];
    attn[row + c0 + 16 + n] = (__bf16)o1[r];
    attn[row + c0 + 32 + n] = (__bf16)o2[r];
    attn[row + c0 + 48 + n] = (__bf16)o3[r];
  }
}

// ---------------------------------------------------------------------------
// Cross-entropy: per-row log-sum-exp over V=32000, nll[row] = lse - logit[tgt]
// ---------------------------------------------------------------------------
__global__ __launch_bounds__(256) void ce_row_kernel(
    const float* __restrict__ logits, const int* __restrict__ targets,
    float* __restrict__ nll) {
  __shared__ float red[256];
  const int row = blockIdx.x;
  const float* lr = logits + (size_t)row * Vk;
  const int tid = threadIdx.x;
  float mx = -3.0e38f;
  for (int i = tid; i < Vk; i += 256) mx = fmaxf(mx, lr[i]);
  red[tid] = mx;
  __syncthreads();
  for (int s = 128; s > 0; s >>= 1) {
    if (tid < s) red[tid] = fmaxf(red[tid], red[tid + s]);
    __syncthreads();
  }
  mx = red[0];
  __syncthreads();
  float sm = 0.f;
  for (int i = tid; i < Vk; i += 256) sm += __expf(lr[i] - mx);
  red[tid] = sm;
  __syncthreads();
  for (int s = 128; s > 0; s >>= 1) {
    if (tid < s) red[tid] += red[tid + s];
    __syncthreads();
  }
  if (tid == 0) nll[row] = (mx + __logf(red[0])) - lr[targets[row]];
}

// Deterministic fixed-order mean of nll -> d_out[B*T*V]
__global__ __launch_bounds__(256) void loss_reduce_kernel(
    const float* __restrict__ nll, float* __restrict__ out_loss) {
  __shared__ float red[256];
  const int tid = threadIdx.x;
  float s = 0.f;
  for (int i = tid; i < Bk * Tk; i += 256) s += nll[i];
  red[tid] = s;
  __syncthreads();
  for (int st = 128; st > 0; st >>= 1) {
    if (tid < st) red[tid] += red[tid + st];
    __syncthreads();
  }
  if (tid == 0) out_loss[0] = red[0] / (float)(Bk * Tk);
}

// ---------------------------------------------------------------------------
extern "C" void kernel_launch(void* const* d_in, const int* in_sizes, int n_in,
                              void* d_out, int out_size, void* d_ws, size_t ws_size,
                              hipStream_t stream) {
  const int* idx        = (const int*)d_in[0];
  const int* targets    = (const int*)d_in[1];
  const float* tok_emb  = (const float*)d_in[2];
  const float* pos_emb  = (const float*)d_in[3];
  const float* Wkw      = (const float*)d_in[4];   // note: Wk before Wq in dict
  const float* Wqw      = (const float*)d_in[5];
  const float* Wvw      = (const float*)d_in[6];
  const float* Wow      = (const float*)d_in[7];
  const float* bo       = (const float*)d_in[8];
  float* out = (float*)d_out;                      // logits [B*T*V] then loss

  char* ws = (char*)d_ws;
  size_t off = 0;
  auto alloc = [&](size_t bytes) -> void* {
    void* p = ws + off;
    off = (off + bytes + 255) & ~(size_t)255;
    return p;
  };
  __bf16* WqT  = (__bf16*)alloc((size_t)Ck * Ck * 2);
  __bf16* WkT  = (__bf16*)alloc((size_t)Ck * Ck * 2);
  __bf16* WvT  = (__bf16*)alloc((size_t)Ck * Ck * 2);
  __bf16* WoT  = (__bf16*)alloc((size_t)Vk * Ck * 2);
  __bf16* xb   = (__bf16*)alloc((size_t)Bk * Tk * Ck * 2);
  __bf16* qb   = (__bf16*)alloc((size_t)Bk * Tk * Ck * 2);
  __bf16* kbuf = (__bf16*)alloc((size_t)Bk * Tk * Ck * 2);
  __bf16* vtb  = (__bf16*)alloc((size_t)Bk * Ck * Tk * 2);
  __bf16* attn = (__bf16*)alloc((size_t)Bk * Tk * Ck * 2);
  float*  nll  = (float*)alloc((size_t)Bk * Tk * 4);

  // 1) weight transpose/convert to bf16 [N][K]
  convert_wT_kernel<<<(Ck * Ck + 255) / 256, 256, 0, stream>>>(Wqw, WqT, Ck, Ck);
  convert_wT_kernel<<<(Ck * Ck + 255) / 256, 256, 0, stream>>>(Wkw, WkT, Ck, Ck);
  convert_wT_kernel<<<(Ck * Ck + 255) / 256, 256, 0, stream>>>(Wvw, WvT, Ck, Ck);
  convert_wT_kernel<<<((size_t)Ck * Vk + 255) / 256, 256, 0, stream>>>(Wow, WoT, Ck, Vk);

  // 2) embeddings -> bf16
  embed_kernel<<<(Bk * Tk * Ck + 255) / 256, 256, 0, stream>>>(idx, tok_emb, pos_emb, xb);

  // 3) Q, K projections (row-major bf16), V projection (transposed [B][C][T])
  {
    const int M = Bk * Tk, N = Ck;
    int waves = (M / 16) * (N / 64);             // 4096
    gemm_rm_kernel<<<waves / 8, 256, 0, stream>>>(xb, WqT, qb, M, Ck, N);
    gemm_rm_kernel<<<waves / 8, 256, 0, stream>>>(xb, WkT, kbuf, M, Ck, N);
    gemm_vt_kernel<<<waves / 8, 256, 0, stream>>>(xb, WvT, vtb);
  }

  // 4) causal attention: one 16-query tile per workgroup
  attn_kernel<<<Bk * (Tk / 16), 256, 0, stream>>>(qb, kbuf, vtb, attn);

  // 5) logits = attn @ Wo + bo  (f32 -> d_out)
  {
    const int M = Bk * Tk, N = Vk;
    int waves = (M / 16) * (N / 64);             // 256000
    gemm_logits_kernel<<<waves / 8, 256, 0, stream>>>(attn, WoT, bo, out);
  }

  // 6) cross-entropy mean -> d_out[B*T*V]
  ce_row_kernel<<<Bk * Tk, 256, 0, stream>>>(out, targets, nll);
  loss_reduce_kernel<<<1, 256, 0, stream>>>(nll, out + (size_t)Bk * Tk * Vk);
}
